// DroplessMoE_68195490726097
// MI455X (gfx1250) — compile-verified, matrix operand order checked by reading, not run on hardware
//
#include <hip/hip_runtime.h>
#include <hip/hip_bf16.h>

typedef __attribute__((ext_vector_type(16))) __bf16 v16bf;
typedef __attribute__((ext_vector_type(8)))  float  v8f;

#define H_DIM 1024
#define F_DIM 2048
#define NEXP  8
#define BM    32      // tokens per block in fused FFN kernel

__device__ inline v8f wmma_bf16(v16bf a, v16bf b, v8f c) {
    // D = A(16x32 bf16) * B(32x16 bf16) + C(16x16 f32)
    return __builtin_amdgcn_wmma_f32_16x16x32_bf16(
        /*neg_a=*/false, a, /*neg_b=*/false, b,
        /*c_mod=*/(short)0, c, /*reuse_a=*/false, /*reuse_b=*/false);
}

// ---------------- fp32 -> bf16 straight conversion ----------------
__global__ __launch_bounds__(256) void cvt_bf16_kernel(
    const float* __restrict__ in, __bf16* __restrict__ out, long n) {
    for (long i = (long)blockIdx.x * blockDim.x + threadIdx.x; i < n;
         i += (long)gridDim.x * blockDim.x) {
        out[i] = (__bf16)in[i];
    }
}

// -------- fp32 [E][K][N] -> bf16 [E][N][K] transpose conversion --------
__global__ __launch_bounds__(256) void cvt_bf16_transpose_kernel(
    const float* __restrict__ in, __bf16* __restrict__ out,
    int Kd, int Nd, long total) {
    for (long i = (long)blockIdx.x * blockDim.x + threadIdx.x; i < total;
         i += (long)gridDim.x * blockDim.x) {
        long k = i % Kd;
        long n = (i / Kd) % Nd;
        long e = i / ((long)Kd * Nd);
        out[i] = (__bf16)in[(e * Kd + k) * Nd + n];
    }
}

// ---------------- router: one wave32 per token ----------------
// scal[0]=sum logz^2, scal[1]=mask-e0 count, scal[2]=mask-e1 count,
// scal[3]=sum top1 w, scal[4]=sum top2 w
__global__ __launch_bounds__(256) void router_kernel(
    const float* __restrict__ x, const float* __restrict__ Wg,
    int* __restrict__ counts, int* __restrict__ lists,
    float* __restrict__ wts, float* __restrict__ scal, int T) {
    int wave = blockIdx.x * (blockDim.x >> 5) + (threadIdx.x >> 5);
    int lane = threadIdx.x & 31;
    if (wave >= T) return;
    const float* xr = x + (size_t)wave * H_DIM;

    float l[NEXP];
#pragma unroll
    for (int e = 0; e < NEXP; ++e) l[e] = 0.f;

    for (int k = lane; k < H_DIM; k += 32) {
        float xv = xr[k];
        const float* wr = Wg + k * NEXP;
#pragma unroll
        for (int e = 0; e < NEXP; ++e) l[e] += xv * wr[e];
    }
#pragma unroll
    for (int e = 0; e < NEXP; ++e) {
#pragma unroll
        for (int s = 16; s > 0; s >>= 1) l[e] += __shfl_xor(l[e], s, 32);
    }
    if (lane == 0) {
        float mx = l[0];
#pragma unroll
        for (int e = 1; e < NEXP; ++e) mx = fmaxf(mx, l[e]);
        float p[NEXP], se = 0.f;
#pragma unroll
        for (int e = 0; e < NEXP; ++e) { p[e] = __expf(l[e] - mx); se += p[e]; }
        float inv = 1.f / se;

        int i0 = 0;
#pragma unroll
        for (int e = 1; e < NEXP; ++e) if (p[e] > p[i0]) i0 = e;
        int i1 = (i0 == 0) ? 1 : 0;
#pragma unroll
        for (int e = 0; e < NEXP; ++e) if (e != i0 && p[e] > p[i1]) i1 = e;
        float p0 = p[i0] * inv, p1 = p[i1] * inv;

        float logz = logf(se) + mx;
        atomicAdd(&scal[0], logz * logz);
        atomicAdd(&scal[1], (i0 == 0 || i1 == 0) ? 1.f : 0.f);
        atomicAdd(&scal[2], (i0 == 1 || i1 == 1) ? 1.f : 0.f);
        atomicAdd(&scal[3], p0);
        atomicAdd(&scal[4], p1);

        int s0 = atomicAdd(&counts[i0], 1);
        lists[i0 * T + s0] = wave;  wts[i0 * T + s0] = p0;
        int s1 = atomicAdd(&counts[i1], 1);
        lists[i1 * T + s1] = wave;  wts[i1 * T + s1] = p1;
    }
}

// ---------------- fused per-expert FFN: WMMA bf16 ----------------
// grid.x = row-blocks, grid.y = expert. 256 threads = 8 wave32 (2x4 tile grid),
// each wave owns TWO 16x16 N-tiles per 128-column chunk (A-fragment reuse).
__global__ __launch_bounds__(256) void moe_ffn_kernel(
    const __bf16* __restrict__ xb,   // [T][H] bf16
    const __bf16* __restrict__ w1t,  // [E][F][H] bf16 (n-major)
    const __bf16* __restrict__ w2t,  // [E][H][F] bf16 (n-major)
    const int* __restrict__ counts, const int* __restrict__ lists,
    const float* __restrict__ wts, float* __restrict__ out, int T) {
    int e = blockIdx.y;
    int cnt = counts[e];
    int row0 = blockIdx.x * BM;
    if (row0 >= cnt) return;

    __shared__ __align__(16) __bf16 xs[BM * H_DIM];   // 64 KB
    __shared__ __align__(16) __bf16 hs[BM * F_DIM];   // 128 KB (CDNA5: 320KB/WGP)
    __shared__ int   tok[BM];
    __shared__ float twt[BM];

    int tid = threadIdx.x;
    if (tid < BM) {
        int r = row0 + tid;
        int tk = 0; float w = 0.f;
        if (r < cnt) { tk = lists[e * T + r]; w = wts[e * T + r]; }
        tok[tid] = tk; twt[tid] = w;
    }
    __syncthreads();

    // async gather of 32 token rows (bf16) straight into LDS (no VGPR bounce):
    // GLOBAL_LOAD_ASYNC_TO_LDS_B128, tracked by ASYNCcnt.
    {
        const int CH = (BM * H_DIM) / 8;  // 8 bf16 per 16B transfer
        for (int i = tid; i < CH; i += 256) {
            int r = i >> 7;              // H/8 = 128 chunks per row
            int c = (i & 127) << 3;
            const __bf16* src = xb + (size_t)tok[r] * H_DIM + c;
            unsigned ldsoff = (unsigned)(size_t)(&xs[r * H_DIM + c]);
            asm volatile("global_load_async_to_lds_b128 %0, %1, off"
                         :: "v"(ldsoff), "v"(src) : "memory");
        }
        asm volatile("s_wait_asynccnt 0x0" ::: "memory");
    }
    __syncthreads();

    int wid = tid >> 5, lane = tid & 31;
    int wm = wid >> 2;                    // 0..1  -> 16-row slice
    int wn = wid & 3;                     // 0..3  -> 16-col slice within chunk
    int m0 = wm * 16;
    int amr = m0 + (lane & 15);           // A fragment row
    int kh  = (lane >> 4) * 16;           // half-lane K offset
    int cmb = m0 + ((lane >> 4) << 3);    // C fragment row base
    int lanN = lane & 15;

    // ---- stage A: h = gelu(X @ W1[e]), stored bf16 in LDS ----
    for (int nc = 0; nc < F_DIM; nc += 128) {
        int n0 = nc + wn * 16;            // first N tile
        int n1 = n0 + 64;                 // second N tile
        v8f acc0 = {0.f, 0.f, 0.f, 0.f, 0.f, 0.f, 0.f, 0.f};
        v8f acc1 = {0.f, 0.f, 0.f, 0.f, 0.f, 0.f, 0.f, 0.f};
        const __bf16* arow  = &xs[amr * H_DIM + kh];
        const __bf16* brow0 = w1t + ((size_t)e * F_DIM + (n0 + lanN)) * H_DIM + kh;
        const __bf16* brow1 = w1t + ((size_t)e * F_DIM + (n1 + lanN)) * H_DIM + kh;
#pragma unroll 4
        for (int k0 = 0; k0 < H_DIM; k0 += 32) {
            v16bf a  = *(const v16bf*)(arow + k0);
            v16bf b0 = *(const v16bf*)(brow0 + k0);
            v16bf b1 = *(const v16bf*)(brow1 + k0);
            acc0 = wmma_bf16(a, b0, acc0);
            acc1 = wmma_bf16(a, b1, acc1);
        }
#pragma unroll
        for (int i = 0; i < 8; ++i) {
            float v0 = acc0[i];
            float v1 = acc1[i];
            float g0 = 0.5f * v0 * (1.0f + erff(v0 * 0.70710678f));  // exact gelu
            float g1 = 0.5f * v1 * (1.0f + erff(v1 * 0.70710678f));
            hs[(cmb + i) * F_DIM + (n0 + lanN)] = (__bf16)g0;
            hs[(cmb + i) * F_DIM + (n1 + lanN)] = (__bf16)g1;
        }
    }
    __syncthreads();

    // ---- stage B: out += weight * (h @ W2[e]) ----
    for (int nc = 0; nc < H_DIM; nc += 128) {
        int n0 = nc + wn * 16;
        int n1 = n0 + 64;
        v8f acc0 = {0.f, 0.f, 0.f, 0.f, 0.f, 0.f, 0.f, 0.f};
        v8f acc1 = {0.f, 0.f, 0.f, 0.f, 0.f, 0.f, 0.f, 0.f};
        const __bf16* arow  = &hs[amr * F_DIM + kh];
        const __bf16* brow0 = w2t + ((size_t)e * H_DIM + (n0 + lanN)) * F_DIM + kh;
        const __bf16* brow1 = w2t + ((size_t)e * H_DIM + (n1 + lanN)) * F_DIM + kh;
#pragma unroll 4
        for (int k0 = 0; k0 < F_DIM; k0 += 32) {
            v16bf a  = *(const v16bf*)(arow + k0);
            v16bf b0 = *(const v16bf*)(brow0 + k0);
            v16bf b1 = *(const v16bf*)(brow1 + k0);
            acc0 = wmma_bf16(a, b0, acc0);
            acc1 = wmma_bf16(a, b1, acc1);
        }
#pragma unroll
        for (int i = 0; i < 8; ++i) {
            int m = cmb + i;
            float w = twt[m];
            size_t orow = (size_t)tok[m] * H_DIM;
            atomicAdd(&out[orow + n0 + lanN], acc0[i] * w);
            atomicAdd(&out[orow + n1 + lanN], acc1[i] * w);
        }
    }
}

// ---------------- loss finalize ----------------
__global__ void finalize_kernel(const float* __restrict__ scal,
                                float* __restrict__ out, int T, long pos) {
    float invT = 1.f / (float)T;
    out[pos]     = scal[0] * invT;  // z_loss
    out[pos + 1] = 2.f * ((scal[1] * invT) * (scal[3] * invT) +
                          (scal[2] * invT) * (scal[4] * invT));  // aux (K=2)
}

extern "C" void kernel_launch(void* const* d_in, const int* in_sizes, int n_in,
                              void* d_out, int out_size, void* d_ws, size_t ws_size,
                              hipStream_t stream) {
    const float* x  = (const float*)d_in[0];
    const float* Wg = (const float*)d_in[1];
    const float* W1 = (const float*)d_in[2];
    const float* W2 = (const float*)d_in[3];
    float* out = (float*)d_out;

    const int T = in_sizes[0] / H_DIM;   // 16384

    // workspace layout
    char* ws = (char*)d_ws;
    size_t off = 0;
    auto take = [&](size_t bytes) -> void* {
        void* p = ws + off;
        off = (off + bytes + 255) & ~(size_t)255;
        return p;
    };
    __bf16* xb   = (__bf16*)take((size_t)T * H_DIM * 2);
    __bf16* w1t  = (__bf16*)take((size_t)NEXP * F_DIM * H_DIM * 2);
    __bf16* w2t  = (__bf16*)take((size_t)NEXP * H_DIM * F_DIM * 2);
    int*    cnts = (int*)  take(NEXP * sizeof(int));
    float*  scal = (float*)take(8 * sizeof(float));
    int*    list = (int*)  take((size_t)NEXP * T * sizeof(int));
    float*  lwts = (float*)take((size_t)NEXP * T * sizeof(float));
    (void)ws_size; (void)n_in;

    // zero output (accumulated atomically) and counters/stat accumulators
    hipMemsetAsync(d_out, 0, (size_t)out_size * sizeof(float), stream);
    hipMemsetAsync(cnts, 0, NEXP * sizeof(int) + 8 * sizeof(float) + 256, stream);

    // precision conversion (once per launch; weights become L2-resident bf16)
    long nx = (long)T * H_DIM;
    cvt_bf16_kernel<<<4096, 256, 0, stream>>>(x, xb, nx);
    long nw = (long)NEXP * H_DIM * F_DIM;
    cvt_bf16_transpose_kernel<<<4096, 256, 0, stream>>>(W1, w1t, H_DIM, F_DIM, nw);
    cvt_bf16_transpose_kernel<<<4096, 256, 0, stream>>>(W2, w2t, F_DIM, H_DIM, nw);

    // routing + losses + dropless expert lists
    router_kernel<<<(T + 7) / 8, 256, 0, stream>>>(x, Wg, cnts, list, lwts, scal, T);

    // fused expert FFN (WMMA bf16, f32 accumulate)
    dim3 grid((T + BM - 1) / BM, NEXP);
    moe_ffn_kernel<<<grid, 256, 0, stream>>>(xb, w1t, w2t, cnts, list, lwts, out, T);

    // scalar losses at tail of output
    finalize_kernel<<<1, 1, 0, stream>>>(scal, out, T, (long)T * H_DIM);
}
